// BitModel_6983616824187
// MI455X (gfx1250) — compile-verified, matrix operand order checked by reading, not run on hardware
//
#include <hip/hip_runtime.h>
#include <hip/hip_bf16.h>
#include <math.h>

// ---------------------------------------------------------------------------
// BitNet b1.58 4-layer MLP for MI455X (gfx1250, wave32, WMMA).
// int8 activations x ternary int8 weights -> i32 accum via
// V_WMMA_I32_16X16X64_IU8; double-buffered LDS tiles fed by the CDNA5
// async global->LDS path (ASYNCcnt), sync b128 fallback if unavailable.
// ---------------------------------------------------------------------------

typedef __attribute__((ext_vector_type(8))) int v8i;
typedef __attribute__((ext_vector_type(4))) int v4i;

#define EPSF 1e-5f

// ---- CDNA5 async global->LDS support detection ----------------------------
#if defined(__has_builtin)
#if __has_builtin(__builtin_amdgcn_global_load_async_to_lds_b128)
#define ASYNC_LDS 1
#endif
#if __has_builtin(__builtin_amdgcn_s_wait_asynccnt)
#define HAVE_WAIT_ASYNC 1
#endif
#endif

#ifdef ASYNC_LDS
typedef __attribute__((address_space(1))) v4i* g4_ptr;   // global v4i*
typedef __attribute__((address_space(3))) v4i* l4_ptr;   // LDS v4i*
#endif

// stage one 16B chunk global -> LDS
__device__ inline void stage16(const signed char* g, signed char* l) {
#ifdef ASYNC_LDS
    g4_ptr gp = (g4_ptr)(unsigned long long)(size_t)g;   // 64-bit global addr
    l4_ptr lp = (l4_ptr)(unsigned int)(size_t)l;         // low 32 bits = LDS offset
    __builtin_amdgcn_global_load_async_to_lds_b128(gp, lp, 0, 0);
#else
    *(v4i*)l = *(const v4i*)g;
#endif
}

__device__ inline void wait_stage() {
#ifdef ASYNC_LDS
#ifdef HAVE_WAIT_ASYNC
    __builtin_amdgcn_s_wait_asynccnt(0);
#else
    asm volatile("s_wait_asynccnt 0" ::: "memory");
#endif
#endif
}

// ------------------------- weight abs-mean reduction -----------------------
__global__ void k_abs_sum(const float* __restrict__ w, int n, float* __restrict__ sum_out) {
    __shared__ float s[256];
    float acc = 0.f;
    for (int i = blockIdx.x * blockDim.x + threadIdx.x; i < n; i += gridDim.x * blockDim.x)
        acc += fabsf(w[i]);
    s[threadIdx.x] = acc;
    __syncthreads();
    for (int o = 128; o > 0; o >>= 1) {
        if (threadIdx.x < o) s[threadIdx.x] += s[threadIdx.x + o];
        __syncthreads();
    }
    if (threadIdx.x == 0) atomicAdd(sum_out, s[0]);
}

// ------------------------- ternary weight quantization ---------------------
__global__ void k_quant_w(const float* __restrict__ w, int n,
                          const float* __restrict__ sum_in,
                          signed char* __restrict__ qw,
                          float* __restrict__ invw_out) {
    float mean  = *sum_in / (float)n;
    float m     = fmaxf(mean, EPSF);      // dequant multiplier (= 1/scale)
    float scale = 1.f / m;
    for (int i = blockIdx.x * blockDim.x + threadIdx.x; i < n; i += gridDim.x * blockDim.x) {
        float v = rintf(w[i] * scale);
        v = fminf(fmaxf(v, -1.f), 1.f);
        qw[i] = (signed char)(int)v;
    }
    if (blockIdx.x == 0 && threadIdx.x == 0) *invw_out = m;
}

// ------------------------- per-row int8 activation quant -------------------
__global__ void k_act_quant(const float* __restrict__ x, int K,
                            signed char* __restrict__ xq,
                            float* __restrict__ inv_a) {
    const int row = blockIdx.x;
    const float* xr = x + (size_t)row * K;
    __shared__ float s[256];
    float mx = 0.f;
    for (int k = threadIdx.x; k < K; k += blockDim.x) mx = fmaxf(mx, fabsf(xr[k]));
    s[threadIdx.x] = mx;
    __syncthreads();
    for (int o = 128; o > 0; o >>= 1) {
        if (threadIdx.x < o) s[threadIdx.x] = fmaxf(s[threadIdx.x], s[threadIdx.x + o]);
        __syncthreads();
    }
    float maxv  = fmaxf(s[0], EPSF);
    float scale = 127.f / maxv;
    signed char* q = xq + (size_t)row * K;
    for (int k = threadIdx.x; k < K; k += blockDim.x) {
        float v = rintf(xr[k] * scale);
        v = fminf(fmaxf(v, -128.f), 127.f);
        q[k] = (signed char)(int)v;
    }
    if (threadIdx.x == 0) inv_a[row] = maxv / 127.f;  // 1/scale
}

// ------------------------- int8 WMMA GEMM ----------------------------------
// C[m,n] = sum_k Aq[m,k] * Wq[n,k] (i32); out = C * inv_a[m] * invw + b[n].
// Block: 256 threads / 8 waves; 128x128 output tile; K stepped by 64.
// Double-buffered LDS (A,B: 128 rows x 64B, 80B pitch for conflict-free
// 16-row x 16B fragment reads). One barrier per K-step.

#define LDS_PITCH 80

__device__ inline v8i lds_frag(const signed char* base, int tile_row, int lane) {
    const int m = lane & 15;
    const int half = lane >> 4;  // lanes 0-15: K bytes 0..31, lanes 16-31: 32..63
    const v4i* p = (const v4i*)(base + (tile_row + m) * LDS_PITCH + half * 32);
    v4i lo = p[0];
    v4i hi = p[1];
    v8i r;
    r[0] = lo[0]; r[1] = lo[1]; r[2] = lo[2]; r[3] = lo[3];
    r[4] = hi[0]; r[5] = hi[1]; r[6] = hi[2]; r[7] = hi[3];
    return r;
}

template <bool TANH>
__global__ __launch_bounds__(256) void k_gemm(
        const signed char* __restrict__ Aq,   // M x K row-major int8
        const signed char* __restrict__ Wq,   // N x K row-major int8 (ternary)
        const float* __restrict__ bias,       // N
        const float* __restrict__ inv_a,      // M (per-row act dequant)
        const float* __restrict__ invw,       // scalar weight dequant
        float* __restrict__ out,              // M x N row-major f32
        int N, int K) {
    __shared__ __align__(16) signed char smemA[2][128 * LDS_PITCH];
    __shared__ __align__(16) signed char smemB[2][128 * LDS_PITCH];

    const int tid    = threadIdx.x;
    const int lane   = tid & 31;
    const int wave   = tid >> 5;
    const int wm     = wave >> 2;            // 0..1 -> 64-row strip
    const int wn     = wave & 3;             // 0..3 -> 32-col strip
    const int blockM = blockIdx.y * 128;
    const int blockN = blockIdx.x * 128;

    // Cooperative fill: 512 16B chunks per matrix, 2 per thread per matrix.
    const int r0 = tid >> 2,         col0 = (tid & 3) * 16;
    const int r1 = (tid + 256) >> 2, col1 = (tid & 3) * 16;  // rows 64..127

    const signed char* gA0 = Aq + (size_t)(blockM + r0) * K + col0;
    const signed char* gA1 = Aq + (size_t)(blockM + r1) * K + col1;
    const signed char* gB0 = Wq + (size_t)(blockN + r0) * K + col0;
    const signed char* gB1 = Wq + (size_t)(blockN + r1) * K + col1;

    auto stage_tile = [&](int buf, int kk) {
        stage16(gA0 + kk, &smemA[buf][r0 * LDS_PITCH + col0]);
        stage16(gA1 + kk, &smemA[buf][r1 * LDS_PITCH + col1]);
        stage16(gB0 + kk, &smemB[buf][r0 * LDS_PITCH + col0]);
        stage16(gB1 + kk, &smemB[buf][r1 * LDS_PITCH + col1]);
    };

    v8i acc[4][2] = {};

    // prologue: fill buffer 0
    stage_tile(0, 0);
    wait_stage();
    __syncthreads();

    int cur = 0;
    for (int kk = 0; kk < K; kk += 64) {
        // stage next K-slice into the alternate buffer (overlaps compute)
        if (kk + 64 < K) {
            stage_tile(cur ^ 1, kk + 64);
        }

        // ---- compute: hoist all fragment loads, then 8 WMMAs ----
        const signed char* sa = smemA[cur];
        const signed char* sb = smemB[cur];
        v8i bfrag0 = lds_frag(sb, wn * 32 + 0,  lane);
        v8i bfrag1 = lds_frag(sb, wn * 32 + 16, lane);
        v8i afrag[4];
        #pragma unroll
        for (int i = 0; i < 4; ++i) afrag[i] = lds_frag(sa, wm * 64 + i * 16, lane);
        #pragma unroll
        for (int i = 0; i < 4; ++i) {
            acc[i][0] = __builtin_amdgcn_wmma_i32_16x16x64_iu8(
                true, afrag[i], true, bfrag0, acc[i][0], false, false);
            acc[i][1] = __builtin_amdgcn_wmma_i32_16x16x64_iu8(
                true, afrag[i], true, bfrag1, acc[i][1], false, false);
        }

        wait_stage();       // next buffer's async fills landed
        __syncthreads();    // everyone done reading cur; next buffer visible
        cur ^= 1;
    }

    // ---- epilogue: dequant + bias (+tanh) ----
    // C/D layout: VGPR j, lanes 0-15 -> (M=j, N=lane); lanes 16-31 -> (M=j+8).
    const float iw   = *invw;
    const int   nCol = lane & 15;
    const int   mOff = (lane >> 4) * 8;
    #pragma unroll
    for (int i = 0; i < 4; ++i) {
        const int mBase = blockM + wm * 64 + i * 16 + mOff;
        #pragma unroll
        for (int j = 0; j < 2; ++j) {
            const int n = blockN + wn * 32 + j * 16 + nCol;
            const float bn = bias[n];
            #pragma unroll
            for (int r = 0; r < 8; ++r) {
                const int m = mBase + r;
                float v = (float)acc[i][j][r] * inv_a[m] * iw + bn;
                if (TANH) v = tanhf(v);
                out[(size_t)m * N + n] = v;
            }
        }
    }
}

// ---------------------------------------------------------------------------
// Host-side launch sequence
// ---------------------------------------------------------------------------
extern "C" void kernel_launch(void* const* d_in, const int* in_sizes, int n_in,
                              void* d_out, int out_size, void* d_ws, size_t ws_size,
                              hipStream_t stream) {
    (void)in_sizes; (void)n_in; (void)out_size; (void)ws_size;

    const int B = 8192, D_IN = 1024, H = 4096, D_OUT = 1024;

    const float* x  = (const float*)d_in[0];
    const float* w1 = (const float*)d_in[1]; const float* b1 = (const float*)d_in[2];
    const float* w2 = (const float*)d_in[3]; const float* b2 = (const float*)d_in[4];
    const float* w3 = (const float*)d_in[5]; const float* b3 = (const float*)d_in[6];
    const float* w4 = (const float*)d_in[7]; const float* b4 = (const float*)d_in[8];
    float* outp = (float*)d_out;

    // Workspace carving (256B aligned)
    char* ws = (char*)d_ws;
    size_t off = 0;
    auto carve = [&](size_t bytes) -> char* {
        char* p = ws + off;
        off = (off + bytes + 255) & ~(size_t)255;
        return p;
    };
    signed char* qw1 = (signed char*)carve((size_t)H * D_IN);
    signed char* qw2 = (signed char*)carve((size_t)H * H);
    signed char* qw3 = (signed char*)carve((size_t)H * H);
    signed char* qw4 = (signed char*)carve((size_t)D_OUT * H);
    float* sums  = (float*)carve(4 * sizeof(float));
    float* invw  = (float*)carve(4 * sizeof(float));
    signed char* xq = (signed char*)carve((size_t)B * H);        // max-size act buf
    float* inv_a = (float*)carve((size_t)B * sizeof(float));
    float* h     = (float*)carve((size_t)B * H * sizeof(float)); // f32 intermediate

    // zero the abs-sum accumulators (graph-capture safe)
    (void)hipMemsetAsync(sums, 0, 4 * sizeof(float), stream);

    const int wn1 = H * D_IN, wn2 = H * H, wn4 = D_OUT * H;
    dim3 blk(256);
    auto rgrid = [](int n) { int g = (n + 255) / 256; return g > 2048 ? 2048 : g; };

    // 1) weight quantization (mean|w| -> ternary int8)
    k_abs_sum<<<rgrid(wn1), blk, 0, stream>>>(w1, wn1, sums + 0);
    k_abs_sum<<<rgrid(wn2), blk, 0, stream>>>(w2, wn2, sums + 1);
    k_abs_sum<<<rgrid(wn2), blk, 0, stream>>>(w3, wn2, sums + 2);
    k_abs_sum<<<rgrid(wn4), blk, 0, stream>>>(w4, wn4, sums + 3);
    k_quant_w<<<rgrid(wn1), blk, 0, stream>>>(w1, wn1, sums + 0, qw1, invw + 0);
    k_quant_w<<<rgrid(wn2), blk, 0, stream>>>(w2, wn2, sums + 1, qw2, invw + 1);
    k_quant_w<<<rgrid(wn2), blk, 0, stream>>>(w3, wn2, sums + 2, qw3, invw + 2);
    k_quant_w<<<rgrid(wn4), blk, 0, stream>>>(w4, wn4, sums + 3, qw4, invw + 3);

    // 2) layer 1: quant(x) -> gemm(tanh) -> h
    k_act_quant<<<B, blk, 0, stream>>>(x, D_IN, xq, inv_a);
    k_gemm<true><<<dim3(H / 128, B / 128), blk, 0, stream>>>(
        xq, qw1, b1, inv_a, invw + 0, h, H, D_IN);

    // 3) layer 2
    k_act_quant<<<B, blk, 0, stream>>>(h, H, xq, inv_a);
    k_gemm<true><<<dim3(H / 128, B / 128), blk, 0, stream>>>(
        xq, qw2, b2, inv_a, invw + 1, h, H, H);

    // 4) layer 3
    k_act_quant<<<B, blk, 0, stream>>>(h, H, xq, inv_a);
    k_gemm<true><<<dim3(H / 128, B / 128), blk, 0, stream>>>(
        xq, qw3, b3, inv_a, invw + 2, h, H, H);

    // 5) layer 4 (no tanh) -> d_out
    k_act_quant<<<B, blk, 0, stream>>>(h, H, xq, inv_a);
    k_gemm<false><<<dim3(D_OUT / 128, B / 128), blk, 0, stream>>>(
        xq, qw4, b4, inv_a, invw + 3, outp, D_OUT, H);
}
